// Attention_17695265259922
// MI455X (gfx1250) — compile-verified, matrix operand order checked by reading, not run on hardware
//
#include <hip/hip_runtime.h>

// Problem dims (fixed by the reference)
#define Bb    2
#define Ss    2048
#define Ee    1024
#define Hh    16
#define Dd    64
#define Mm    (Bb * Ss)        // 4096 rows
#define SCALE 0.125f           // 1/sqrt(64)

typedef __bf16 bf16;
typedef __attribute__((ext_vector_type(16))) __bf16 v16bf;
typedef __attribute__((ext_vector_type(8)))  __bf16 v8bf;
typedef __attribute__((ext_vector_type(8)))  float  v8f;

// ---------------------------------------------------------------------------
// WMMA helper: D = A(16x32 bf16) * B(32x16 bf16) + C(16x16 f32)
// ---------------------------------------------------------------------------
__device__ __forceinline__ v8f wmma_bf16(v16bf a, v16bf b, v8f c) {
  return __builtin_amdgcn_wmma_f32_16x16x32_bf16(false, a, false, b,
                                                 (short)0, c, false, false);
}

// Load one 16x32 bf16 fragment (A layout; also B layout when the matrix is
// stored N-major, i.e. "transposed").  ISA 7.12.2: lane L holds row (L&15),
// halves[0..7] = K kb+0..7, halves[8..15] = K kb+16..23, kb = (L>>4)*8.
// Two contiguous 16-byte loads per lane.
__device__ __forceinline__ v16bf load_frag(const bf16* __restrict__ base,
                                           int stride, int lane) {
  const int row = lane & 15;
  const int kb  = (lane >> 4) << 3;
  const bf16* p = base + row * stride + kb;
  v8bf lo = *reinterpret_cast<const v8bf*>(p);
  v8bf hi = *reinterpret_cast<const v8bf*>(p + 16);
  v16bf r;
#pragma unroll
  for (int i = 0; i < 8; ++i) { r[i] = lo[i]; r[i + 8] = hi[i]; }
  return r;
}

// ---------------------------------------------------------------------------
// Phase 0: conversions
// ---------------------------------------------------------------------------
__global__ void __launch_bounds__(256) cvt_f32_bf16(const float* __restrict__ src,
                                                    bf16* __restrict__ dst, int n) {
  int i = blockIdx.x * blockDim.x + threadIdx.x;
  if (i < n) dst[i] = (bf16)src[i];
}

// W [K=1024][N=1024] f32  ->  Wt [N][K] bf16 (so B-frag loads are contiguous)
__global__ void __launch_bounds__(256) cvt_w_transpose(const float* __restrict__ W,
                                                       bf16* __restrict__ Wt) {
  int i = blockIdx.x * blockDim.x + threadIdx.x;   // 0 .. 1M-1
  int k = i >> 10;
  int n = i & 1023;
  Wt[n * Ee + k] = (bf16)W[k * Ee + n];
}

// ---------------------------------------------------------------------------
// Phase 1: QKV projection.  One wave = one 64x64 output tile.
//   grid.z-free flat decomposition: 3 matrices * 64 mtiles * 16 ntiles = 3072 waves.
//   Q,K stored row-major [m][n] bf16;  V stored transposed Vt[n][m] bf16.
//   For Q,K: the 64-col n-tile == one head, so row sum-of-squares (norms) are
//   reduced here with shfl_xor and written to qn/kn [m][H].
// ---------------------------------------------------------------------------
__global__ void __launch_bounds__(128) gemm_qkv_kernel(
    const bf16* __restrict__ X,                       // [4096][1024]
    const bf16* __restrict__ WqT, const bf16* __restrict__ WkT,
    const bf16* __restrict__ WvT,                     // each [1024][1024] N-major
    bf16* __restrict__ Q, bf16* __restrict__ K, bf16* __restrict__ Vt,
    float* __restrict__ qn, float* __restrict__ kn) {
  const int lane = threadIdx.x & 31;
  const int wave = threadIdx.x >> 5;
  int gw  = blockIdx.x * 4 + wave;                    // 0..3071
  int mat = gw / (64 * 16);
  int rem = gw % (64 * 16);
  int mt  = rem / 16;
  int nt  = rem % 16;                                 // == head for Q/K
  const bf16* Wt = (mat == 0) ? WqT : (mat == 1) ? WkT : WvT;
  const int mBase = mt * 64, nBase = nt * 64;

  v8f acc[4][4];
#pragma unroll
  for (int i = 0; i < 4; ++i)
#pragma unroll
    for (int j = 0; j < 4; ++j)
#pragma unroll
      for (int e = 0; e < 8; ++e) acc[i][j][e] = 0.f;

  for (int kk = 0; kk < Ee; kk += 32) {
    v16bf a[4], bb[4];
#pragma unroll
    for (int i = 0; i < 4; ++i)
      a[i] = load_frag(X + (mBase + i * 16) * Ee + kk, Ee, lane);
#pragma unroll
    for (int j = 0; j < 4; ++j)
      bb[j] = load_frag(Wt + (nBase + j * 16) * Ee + kk, Ee, lane);
    if (kk + 32 < Ee) {   // speculative prefetch of next A k-slice
      __builtin_prefetch(X + mBase * Ee + kk + 32, 0, 0);
    }
#pragma unroll
    for (int i = 0; i < 4; ++i)
#pragma unroll
      for (int j = 0; j < 4; ++j)
        acc[i][j] = wmma_bf16(a[i], bb[j], acc[i][j]);
  }

  const int n   = lane & 15;
  const int hi8 = (lane >> 4) << 3;

  if (mat == 2) {
    // V transposed store: per (i,j) lane holds 8 consecutive m -> one b128.
#pragma unroll
    for (int i = 0; i < 4; ++i)
#pragma unroll
      for (int j = 0; j < 4; ++j) {
        v8bf pack;
#pragma unroll
        for (int r = 0; r < 8; ++r) pack[r] = (bf16)acc[i][j][r];
        *reinterpret_cast<v8bf*>(
            Vt + (nBase + j * 16 + n) * Mm + mBase + i * 16 + hi8) = pack;
      }
  } else {
    bf16*  O   = (mat == 0) ? Q : K;
    float* NRM = (mat == 0) ? qn : kn;
#pragma unroll
    for (int i = 0; i < 4; ++i)
#pragma unroll
      for (int j = 0; j < 4; ++j)
#pragma unroll
        for (int r = 0; r < 8; ++r)
          O[(mBase + i * 16 + hi8 + r) * Ee + nBase + j * 16 + n] =
              (bf16)acc[i][j][r];
    // per-row sum of squares over this head's 64 columns
#pragma unroll
    for (int i = 0; i < 4; ++i) {
      float s[8];
#pragma unroll
      for (int r = 0; r < 8; ++r) {
        float t = 0.f;
#pragma unroll
        for (int j = 0; j < 4; ++j) t += acc[i][j][r] * acc[i][j][r];
        s[r] = t;
      }
#pragma unroll
      for (int msk = 1; msk < 16; msk <<= 1)
#pragma unroll
        for (int r = 0; r < 8; ++r) s[r] += __shfl_xor(s[r], msk, 32);
      if (n == 0) {
#pragma unroll
        for (int r = 0; r < 8; ++r)
          NRM[(mBase + i * 16 + hi8 + r) * Hh + nt] = s[r];
      }
    }
  }
}

// ---------------------------------------------------------------------------
// Phase 2: fused RBF attention.  One wave = (b, h, 16-query tile).
//   Keys processed in 32-chunks starting at the diagonal chunk (strict-upper
//   mask kills everything before it).  Scores exp'ed in registers, bounced
//   through a private LDS slice to A-fragment layout, then P @ V via WMMA.
// ---------------------------------------------------------------------------
__global__ void __launch_bounds__(128) attn_kernel(
    const bf16* __restrict__ Q, const bf16* __restrict__ K,
    const bf16* __restrict__ Vt, const float* __restrict__ qn,
    const float* __restrict__ kn, const float* __restrict__ gamma,
    bf16* __restrict__ O) {
  __shared__ __align__(16) bf16 ldsP[4][16 * 32];
  const int lane = threadIdx.x & 31;
  const int wave = threadIdx.x >> 5;
  int gw  = blockIdx.x * 4 + wave;                    // 0..4095
  int b   = gw / (Hh * (Ss / 16));
  int rem = gw % (Hh * (Ss / 16));
  int h   = rem / (Ss / 16);
  int qt  = rem % (Ss / 16);
  const int sBase   = qt * 16;                        // query pos within batch
  const int rowBase = b * Ss + sBase;                 // global row
  const int n   = lane & 15;
  const int hi8 = (lane >> 4) << 3;

  const float g = gamma[h] * SCALE;

  // Q A-fragments for d=0..31 and d=32..63 (live across the whole key loop)
  v16bf aQ0 = load_frag(Q + rowBase * Ee + h * Dd + 0,  Ee, lane);
  v16bf aQ1 = load_frag(Q + rowBase * Ee + h * Dd + 32, Ee, lane);

  float qnr[8];
#pragma unroll
  for (int r = 0; r < 8; ++r) qnr[r] = qn[(rowBase + hi8 + r) * Hh + h];

  v8f oacc[4];
#pragma unroll
  for (int j = 0; j < 4; ++j)
#pragma unroll
    for (int e = 0; e < 8; ++e) oacc[j][e] = 0.f;

  bf16* myLds = ldsP[wave];
  const int kb0 = sBase & ~31;                        // first chunk with t > s

  for (int kb = kb0; kb < Ss; kb += 32) {
#pragma unroll
    for (int c = 0; c < 2; ++c) {
      const int t0 = kb + c * 16;
      v16bf bK0 = load_frag(K + (b * Ss + t0) * Ee + h * Dd + 0,  Ee, lane);
      v16bf bK1 = load_frag(K + (b * Ss + t0) * Ee + h * Dd + 32, Ee, lane);
      v8f s;
#pragma unroll
      for (int e = 0; e < 8; ++e) s[e] = 0.f;
      s = wmma_bf16(aQ0, bK0, s);
      s = wmma_bf16(aQ1, bK1, s);
      const float knn = kn[(b * Ss + t0 + n) * Hh + h];
      const int   t   = t0 + n;
#pragma unroll
      for (int r = 0; r < 8; ++r) {
        const int   srow = sBase + hi8 + r;
        float dist = qnr[r] - 2.f * s[r] + knn;
        dist = fmaxf(dist, 0.f);
        const float p = (t > srow) ? __expf(-g * dist) : 0.f;
        myLds[(hi8 + r) * 32 + c * 16 + n] = (bf16)p;
      }
    }
    // Re-read the 16x32 score tile in A-fragment layout (intra-wave LDS RAW;
    // compiler inserts s_wait_dscnt).
    v16bf aP;
    {
      const bf16* p = myLds + (lane & 15) * 32 + ((lane >> 4) << 3);
      v8bf lo = *reinterpret_cast<const v8bf*>(p);
      v8bf hi = *reinterpret_cast<const v8bf*>(p + 16);
#pragma unroll
      for (int i = 0; i < 8; ++i) { aP[i] = lo[i]; aP[i + 8] = hi[i]; }
    }
#pragma unroll
    for (int j = 0; j < 4; ++j) {
      v16bf bV = load_frag(Vt + (h * Dd + j * 16) * Mm + b * Ss + kb, Mm, lane);
      oacc[j] = wmma_bf16(aP, bV, oacc[j]);
    }
  }

#pragma unroll
  for (int j = 0; j < 4; ++j)
#pragma unroll
    for (int r = 0; r < 8; ++r)
      O[(rowBase + hi8 + r) * Ee + h * Dd + j * 16 + n] = (bf16)oacc[j][r];
}

// ---------------------------------------------------------------------------
// Phase 3: output projection, f32 result straight to d_out.
// ---------------------------------------------------------------------------
__global__ void __launch_bounds__(128) gemm_out_kernel(
    const bf16* __restrict__ X,          // Obf [4096][1024]
    const bf16* __restrict__ WoT,        // [1024][1024] N-major
    float* __restrict__ out) {
  const int lane = threadIdx.x & 31;
  const int wave = threadIdx.x >> 5;
  int gw = blockIdx.x * 4 + wave;                     // 0..1023
  int mt = gw / 16;
  int nt = gw % 16;
  const int mBase = mt * 64, nBase = nt * 64;

  v8f acc[4][4];
#pragma unroll
  for (int i = 0; i < 4; ++i)
#pragma unroll
    for (int j = 0; j < 4; ++j)
#pragma unroll
      for (int e = 0; e < 8; ++e) acc[i][j][e] = 0.f;

  for (int kk = 0; kk < Ee; kk += 32) {
    v16bf a[4], bb[4];
#pragma unroll
    for (int i = 0; i < 4; ++i)
      a[i] = load_frag(X + (mBase + i * 16) * Ee + kk, Ee, lane);
#pragma unroll
    for (int j = 0; j < 4; ++j)
      bb[j] = load_frag(WoT + (nBase + j * 16) * Ee + kk, Ee, lane);
#pragma unroll
    for (int i = 0; i < 4; ++i)
#pragma unroll
      for (int j = 0; j < 4; ++j)
        acc[i][j] = wmma_bf16(a[i], bb[j], acc[i][j]);
  }

  const int n   = lane & 15;
  const int hi8 = (lane >> 4) << 3;
#pragma unroll
  for (int i = 0; i < 4; ++i)
#pragma unroll
    for (int j = 0; j < 4; ++j)
#pragma unroll
      for (int r = 0; r < 8; ++r)
        out[(mBase + i * 16 + hi8 + r) * Ee + nBase + j * 16 + n] =
            acc[i][j][r];
}

// ---------------------------------------------------------------------------
// Host orchestration
// ---------------------------------------------------------------------------
extern "C" void kernel_launch(void* const* d_in, const int* in_sizes, int n_in,
                              void* d_out, int out_size, void* d_ws,
                              size_t ws_size, hipStream_t stream) {
  const float* q     = (const float*)d_in[0];   // [2,2048,1024]
  const float* Wq    = (const float*)d_in[1];   // [1024,1024]
  const float* Wk    = (const float*)d_in[2];
  const float* Wv    = (const float*)d_in[3];
  const float* Wo    = (const float*)d_in[4];
  const float* gamma = (const float*)d_in[5];   // [16]
  float* out = (float*)d_out;                   // [2,2048,1024]

  // Workspace carve-up (~48.5 MB total)
  char* ws = (char*)d_ws;
  bf16* Xbf = (bf16*)ws; ws += (size_t)Mm * Ee * 2;       // 8 MB
  bf16* WqT = (bf16*)ws; ws += (size_t)Ee * Ee * 2;       // 2 MB
  bf16* WkT = (bf16*)ws; ws += (size_t)Ee * Ee * 2;
  bf16* WvT = (bf16*)ws; ws += (size_t)Ee * Ee * 2;
  bf16* WoT = (bf16*)ws; ws += (size_t)Ee * Ee * 2;
  bf16* Qb  = (bf16*)ws; ws += (size_t)Mm * Ee * 2;       // 8 MB
  bf16* Kb  = (bf16*)ws; ws += (size_t)Mm * Ee * 2;
  bf16* Vt  = (bf16*)ws; ws += (size_t)Mm * Ee * 2;       // [n][m]
  bf16* Ob  = (bf16*)ws; ws += (size_t)Mm * Ee * 2;
  float* qn = (float*)ws; ws += (size_t)Mm * Hh * 4;      // 256 KB
  float* kn = (float*)ws; ws += (size_t)Mm * Hh * 4;

  // Phase 0: precision conversion / weight transposition
  cvt_f32_bf16<<<(Mm * Ee + 255) / 256, 256, 0, stream>>>(q, Xbf, Mm * Ee);
  cvt_w_transpose<<<(Ee * Ee) / 256, 256, 0, stream>>>(Wq, WqT);
  cvt_w_transpose<<<(Ee * Ee) / 256, 256, 0, stream>>>(Wk, WkT);
  cvt_w_transpose<<<(Ee * Ee) / 256, 256, 0, stream>>>(Wv, WvT);
  cvt_w_transpose<<<(Ee * Ee) / 256, 256, 0, stream>>>(Wo, WoT);

  // Phase 1: QKV projection + norms (3 * 64 * 16 = 3072 waves -> 768 blocks)
  gemm_qkv_kernel<<<768, 128, 0, stream>>>(Xbf, WqT, WkT, WvT, Qb, Kb, Vt, qn,
                                           kn);

  // Phase 2: fused RBF attention (2*16*128 = 4096 waves -> 1024 blocks)
  attn_kernel<<<1024, 128, 0, stream>>>(Qb, Kb, Vt, qn, kn, gamma, Ob);

  // Phase 3: output projection (64*16 = 1024 waves -> 256 blocks)
  gemm_out_kernel<<<256, 128, 0, stream>>>(Ob, WoT, out);
}
